// Mamba2Layer_3521873183084
// MI455X (gfx1250) — matrix operand order repaired and checked by reading, not compiled, not run on hardware
//
#include <hip/hip_runtime.h>
#include <hip/hip_bf16.h>

// ---------------------------------------------------------------------------
// Problem dims (fixed by reference)
// ---------------------------------------------------------------------------
#define BATCH   2
#define SEQ     4096
#define DMODEL  1024
#define DSTATE  64
#define DINNER  2048
#define MROWS   (BATCH * SEQ)   // 8192 token rows

typedef __attribute__((ext_vector_type(16))) __bf16 v16bf;
typedef __attribute__((ext_vector_type(8)))  __bf16 v8bf;
typedef __attribute__((ext_vector_type(4)))  __bf16 v4bf;
typedef __attribute__((ext_vector_type(8)))  float  v8f;
typedef __attribute__((ext_vector_type(4)))  float  v4f;

union V16 { v16bf v; v8bf h[2]; };

__device__ __forceinline__ float silu_f(float v) {
    return v / (1.0f + __expf(-v));
}
__device__ __forceinline__ float softplus_f(float v) {
    return (v > 20.0f) ? v : log1pf(__expf(v));
}

// ---------------------------------------------------------------------------
// CDNA5 async global->LDS copy (16B per lane), tracked by ASYNCcnt.
// Generic pointer low 32 bits == raw LDS byte offset (flat aperture rule).
// Async loads complete in order, so waiting ASYNCcnt<=N retires oldest groups.
// ---------------------------------------------------------------------------
__device__ __forceinline__ void async_cp16(__bf16* lds_dst, const __bf16* gsrc) {
    unsigned loff = (unsigned)(unsigned long long)lds_dst;
    asm volatile("global_load_async_to_lds_b128 %0, %1, off"
                 :: "v"(loff), "v"(gsrc) : "memory");
}
__device__ __forceinline__ void async_wait0() {
    asm volatile("s_wait_asynccnt 0" ::: "memory");
}
__device__ __forceinline__ void async_wait4() {
    asm volatile("s_wait_asynccnt 4" ::: "memory");
}
__device__ __forceinline__ void async_wait3() {
    asm volatile("s_wait_asynccnt 3" ::: "memory");
}

// ---------------------------------------------------------------------------
// One-time precision conversion: f32 -> bf16 elementwise (for x)
// ---------------------------------------------------------------------------
__global__ void cvt_f32_bf16_kernel(const float* __restrict__ src,
                                    __bf16* __restrict__ dst, size_t n4) {
    size_t i = (size_t)blockIdx.x * blockDim.x + threadIdx.x;
    if (i < n4) {
        v4f a = *(const v4f*)(src + i * 4);
        v4bf b;
        b[0] = (__bf16)a[0]; b[1] = (__bf16)a[1];
        b[2] = (__bf16)a[2]; b[3] = (__bf16)a[3];
        *(v4bf*)(dst + i * 4) = b;
    }
}

// ---------------------------------------------------------------------------
// One-time weight transpose+convert: W[K,N] f32 -> Wt[N,K] bf16 (32x32 tiles)
// ---------------------------------------------------------------------------
__global__ void __launch_bounds__(256) wtrans_kernel(
    const float* __restrict__ W, __bf16* __restrict__ Wt, int K, int N) {
    __shared__ float t[32 * 33];
    const int k0 = blockIdx.x * 32, n0 = blockIdx.y * 32;
    const int c = threadIdx.x & 31, r0 = threadIdx.x >> 5;
    #pragma unroll
    for (int p = 0; p < 4; ++p) {
        int r = r0 + p * 8;
        t[r * 33 + c] = W[(size_t)(k0 + r) * N + n0 + c];
    }
    __syncthreads();
    #pragma unroll
    for (int p = 0; p < 4; ++p) {
        int rr = r0 + p * 8;   // n index
        Wt[(size_t)(n0 + rr) * K + k0 + c] = (__bf16)t[c * 33 + rr];
    }
}

// ---------------------------------------------------------------------------
// Stage 256 rows x 16 k of A + 256 cols x 16 k of B (bf16) into LDS via async DMA.
// 4 async b128 instructions per thread-group pass == 4 per wave (in-order).
// ---------------------------------------------------------------------------
__device__ __forceinline__ void stage_tiles128(
    __bf16* Asb, __bf16* Bsb,
    const __bf16* __restrict__ Abf, const __bf16* __restrict__ Wt,
    int rowBase, int nBase, int K, int k0, int tid)
{
    #pragma unroll
    for (int i = 0; i < 2; ++i) {
        int v   = tid + i * 256;        // 0..511
        int row = v >> 1;               // 0..255
        int kc  = (v & 1) * 8;          // 0 or 8 (x8 bf16 = 16B)
        async_cp16(Asb + row * 32 + kc,
                   Abf + (size_t)(rowBase + row) * K + k0 + kc);
    }
    #pragma unroll
    for (int i = 0; i < 2; ++i) {
        int v  = tid + i * 256;
        int nc = v >> 1;
        int kc = (v & 1) * 8;
        async_cp16(Bsb + nc * 32 + kc,
                   Wt + (size_t)(nBase + nc) * K + k0 + kc);
    }
}

// ---------------------------------------------------------------------------
// Generic 128x128 WMMA GEMM, C = A[M,K](bf16) * Wt[N,K](bf16)^T, f32 accum.
// Double-buffered async global->LDS pipeline: DMA for step k+1 overlaps the
// 8 WMMAs/wave of step k.
// EPI 0: +b_in, SiLU, split -> xinB / gateB (bf16)
// EPI 1: +b_dt, softplus, deterministic row partial sums
// EPI 2: +b_out + x*D -> out (f32)
// Block: 256 threads (8 waves as 2x4), wave tile 64x32 = 4x2 fragments.
// ---------------------------------------------------------------------------
template <int EPI>
__global__ void __launch_bounds__(256) gemm_tile_kernel(
    const __bf16* __restrict__ Abf,    // [M,K] bf16
    const __bf16* __restrict__ Wt,     // [N,K] bf16 (pre-transposed)
    const float*  __restrict__ bias,   // [N]
    int K, int N,
    __bf16* __restrict__ xinB,         // EPI 0
    __bf16* __restrict__ gateB,        // EPI 0
    float*  __restrict__ mdpartial,    // EPI 1: [MROWS, N/128]
    const float* __restrict__ xres,    // EPI 2
    const float* __restrict__ Dvec,    // EPI 2
    float*  __restrict__ outp)         // EPI 2
{
    __shared__ __bf16 As[2][128 * 32];  // ping-pong [row][k]  2x8 KB
    __shared__ __bf16 Bs[2][128 * 32];  // ping-pong [col][k]  2x8 KB
    __shared__ float  mdpart[128 * 4];  // EPI 1 cross-wave row sums

    const int tid    = threadIdx.x;
    const int wave   = tid >> 5;
    const int lane   = tid & 31;
    const int half   = lane >> 4;
    const int l16    = lane & 15;
    const int wave_m = wave >> 2;      // 0..1  (64 rows each)
    const int wave_n = wave & 3;       // 0..3  (32 cols each)

    const int rowBase = blockIdx.x * 128;
    const int nBase   = blockIdx.y * 128;

    v8f zf = {};
    v8f acc[4][2];
    for (int i = 0; i < 4; ++i)
        for (int j = 0; j < 2; ++j) acc[i][j] = zf;

    const int nsteps = K >> 5;
    stage_tiles128(As[0], Bs[0], Abf, Wt, rowBase, nBase, K, 0, tid);

    for (int s = 0; s < nsteps; ++s) {
        const int cur = s & 1;
        if (s + 1 < nsteps) {
            // prefetch next K-slice into the other buffer (protected by the
            // end-of-iteration barrier of step s-1)
            stage_tiles128(As[cur ^ 1], Bs[cur ^ 1], Abf, Wt,
                           rowBase, nBase, K, (s + 1) * 32, tid);
            async_wait4();   // oldest group (buffer `cur`) retired; next in flight
        } else {
            async_wait0();
        }
        __syncthreads();

        // ---- fragments (ISA 7.12.2 bf16 layouts) ----
        const __bf16* Ab = As[cur];
        const __bf16* Bb = Bs[cur];
        V16 af[4];
        #pragma unroll
        for (int fm = 0; fm < 4; ++fm) {
            int r = wave_m * 64 + fm * 16 + l16;
            const __bf16* p = Ab + r * 32 + 8 * half;
            af[fm].h[0] = *(const v8bf*)(p);        // K = 8h .. 8h+7
            af[fm].h[1] = *(const v8bf*)(p + 16);   // K = 16+8h .. 23+8h
        }
        V16 bf[2];
        #pragma unroll
        for (int fn = 0; fn < 2; ++fn) {
            int c = wave_n * 32 + fn * 16 + l16;
            bf[fn].h[0] = *(const v8bf*)(Bb + c * 32 + 8 * half);      // K = 8h .. 8h+7
            bf[fn].h[1] = *(const v8bf*)(Bb + c * 32 + 8 * half + 16); // K = 16+8h .. 23+8h
        }
        #pragma unroll
        for (int fm = 0; fm < 4; ++fm)
            #pragma unroll
            for (int fn = 0; fn < 2; ++fn)
                acc[fm][fn] = __builtin_amdgcn_wmma_f32_16x16x32_bf16(
                    false, af[fm].v, false, bf[fn].v,
                    (short)0, acc[fm][fn], false, false);
        __syncthreads();
    }

    // ---- epilogues ----
    if constexpr (EPI == 0) {
        #pragma unroll
        for (int fm = 0; fm < 4; ++fm)
            #pragma unroll
            for (int fn = 0; fn < 2; ++fn)
                #pragma unroll
                for (int e = 0; e < 8; ++e) {
                    int row = rowBase + wave_m * 64 + fm * 16 + e + 8 * half;
                    int col = nBase + wave_n * 32 + fn * 16 + l16;
                    float s = silu_f(acc[fm][fn][e] + bias[col]);
                    if (col < DINNER)
                        xinB[(size_t)row * DINNER + col] = (__bf16)s;
                    else
                        gateB[(size_t)row * DINNER + (col - DINNER)] = (__bf16)s;
                }
    } else if constexpr (EPI == 1) {
        #pragma unroll
        for (int fm = 0; fm < 4; ++fm) {
            float rs[8];
            #pragma unroll
            for (int e = 0; e < 8; ++e) rs[e] = 0.0f;
            #pragma unroll
            for (int fn = 0; fn < 2; ++fn)
                #pragma unroll
                for (int e = 0; e < 8; ++e) {
                    int col = nBase + wave_n * 32 + fn * 16 + l16;
                    rs[e] += softplus_f(acc[fm][fn][e] + bias[col]);
                }
            // reduce across the 16 lanes of each half (columns of this row)
            #pragma unroll
            for (int e = 0; e < 8; ++e) {
                #pragma unroll
                for (int m = 1; m < 16; m <<= 1)
                    rs[e] += __shfl_xor(rs[e], m, 32);
            }
            if (l16 == 0) {
                #pragma unroll
                for (int e = 0; e < 8; ++e) {
                    int rl = wave_m * 64 + fm * 16 + e + 8 * half;
                    mdpart[rl * 4 + wave_n] = rs[e];
                }
            }
        }
        __syncthreads();
        if (tid < 128) {
            float s = mdpart[tid * 4 + 0] + mdpart[tid * 4 + 1] +
                      mdpart[tid * 4 + 2] + mdpart[tid * 4 + 3];
            mdpartial[(size_t)(rowBase + tid) * (DMODEL / 128) + blockIdx.y] = s;
        }
    } else { // EPI == 2
        #pragma unroll
        for (int fm = 0; fm < 4; ++fm)
            #pragma unroll
            for (int fn = 0; fn < 2; ++fn)
                #pragma unroll
                for (int e = 0; e < 8; ++e) {
                    int row = rowBase + wave_m * 64 + fm * 16 + e + 8 * half;
                    int col = nBase + wave_n * 32 + fn * 16 + l16;
                    float v = acc[fm][fn][e] + bias[col] +
                              xres[(size_t)row * DMODEL + col] * Dvec[col];
                    outp[(size_t)row * DMODEL + col] = v;
                }
    }
}

// ---------------------------------------------------------------------------
// bx = xin(bf16) @ W_xp + b_xp ; M=8192, K=2048, N=64. Tile 128x64, 8 waves 4x2.
// Same double-buffered async pipeline (3 async instructions per stage group).
// ---------------------------------------------------------------------------
__global__ void __launch_bounds__(256) gemm_xp_kernel(
    const __bf16* __restrict__ Abf, const __bf16* __restrict__ Wt /*[64,2048]*/,
    const float* __restrict__ bias, float* __restrict__ bx)
{
    __shared__ __bf16 As[2][128 * 32];   // [row][k]
    __shared__ __bf16 Bs[2][64 * 32];    // [col][k]

    const int tid    = threadIdx.x;
    const int wave   = tid >> 5;
    const int lane   = tid & 31;
    const int half   = lane >> 4;
    const int l16    = lane & 15;
    const int wave_m = wave >> 1;     // 0..3 (32 rows each)
    const int wave_n = wave & 1;      // 0..1 (32 cols each)
    const int rowBase = blockIdx.x * 128;

    v8f zf = {};
    v8f acc[2][2];
    for (int i = 0; i < 2; ++i)
        for (int j = 0; j < 2; ++j) acc[i][j] = zf;

    auto stage = [&](int buf, int k0) {
        #pragma unroll
        for (int i = 0; i < 2; ++i) {
            int v   = tid + i * 256;
            int row = v >> 1;
            int kc  = (v & 1) * 8;
            async_cp16(&As[buf][row * 32 + kc],
                       Abf + (size_t)(rowBase + row) * DINNER + k0 + kc);
        }
        {
            int nc = tid >> 2;            // 0..63
            int kc = (tid & 3) * 8;       // 0..24
            async_cp16(&Bs[buf][nc * 32 + kc],
                       Wt + (size_t)nc * DINNER + k0 + kc);
        }
    };

    const int nsteps = DINNER >> 5;
    stage(0, 0);

    for (int s = 0; s < nsteps; ++s) {
        const int cur = s & 1;
        if (s + 1 < nsteps) {
            stage(cur ^ 1, (s + 1) * 32);
            async_wait3();
        } else {
            async_wait0();
        }
        __syncthreads();

        V16 af[2], bf[2];
        #pragma unroll
        for (int fm = 0; fm < 2; ++fm) {
            int r = wave_m * 32 + fm * 16 + l16;
            const __bf16* p = &As[cur][r * 32 + 8 * half];
            af[fm].h[0] = *(const v8bf*)(p);
            af[fm].h[1] = *(const v8bf*)(p + 16);
        }
        #pragma unroll
        for (int fn = 0; fn < 2; ++fn) {
            int c = wave_n * 32 + fn * 16 + l16;
            bf[fn].h[0] = *(const v8bf*)(&Bs[cur][c * 32 + 8 * half]);
            bf[fn].h[1] = *(const v8bf*)(&Bs[cur][c * 32 + 8 * half + 16]);
        }
        #pragma unroll
        for (int fm = 0; fm < 2; ++fm)
            #pragma unroll
            for (int fn = 0; fn < 2; ++fn)
                acc[fm][fn] = __builtin_amdgcn_wmma_f32_16x16x32_bf16(
                    false, af[fm].v, false, bf[fn].v,
                    (short)0, acc[fm][fn], false, false);
        __syncthreads();
    }

    #pragma unroll
    for (int fm = 0; fm < 2; ++fm)
        #pragma unroll
        for (int fn = 0; fn < 2; ++fn)
            #pragma unroll
            for (int e = 0; e < 8; ++e) {
                int row = rowBase + wave_m * 32 + fm * 16 + e + 8 * half;
                int col = wave_n * 32 + fn * 16 + l16;
                bx[(size_t)row * DSTATE + col] = acc[fm][fn][e] + bias[col];
            }
}

// ---------------------------------------------------------------------------
// md[row] = sum over 8 column-block partials (deterministic reduction)
// ---------------------------------------------------------------------------
__global__ void reduce_md_kernel(const float* __restrict__ part, float* __restrict__ md)
{
    int row = blockIdx.x * blockDim.x + threadIdx.x;
    if (row < MROWS) {
        float s = 0.0f;
        #pragma unroll
        for (int j = 0; j < DMODEL / 128; ++j) s += part[(size_t)row * (DMODEL / 128) + j];
        md[row] = s;
    }
}

// ---------------------------------------------------------------------------
// Chunked selective scan: 2 blocks (one per batch) x 64 threads (one per state)
// ---------------------------------------------------------------------------
__global__ void __launch_bounds__(64) scan_kernel(
    const float* __restrict__ bx, const float* __restrict__ md,
    const float* __restrict__ A_log, float* __restrict__ ssum)
{
    __shared__ float buf[64 * 65];  // [step_in_chunk][state], padded
    const int b = blockIdx.x;
    const int n = threadIdx.x;
    const float An   = -__expf(A_log[n]);
    const float invD = 1.0f / (float)DMODEL;

    float state = 0.0f;
    for (int s0 = 0; s0 < SEQ; s0 += 64) {
        #pragma unroll 4
        for (int i = 0; i < 64; ++i) {
            int s = s0 + i;
            float mean = md[b * SEQ + s] * invD;
            float a = __expf(An * mean);
            state = a * state + bx[((size_t)(b * SEQ + s)) * DSTATE + n];
            buf[i * 65 + n] = state;
        }
        __syncthreads();
        float sum = 0.0f;
        #pragma unroll 8
        for (int i = 0; i < 64; ++i) sum += buf[n * 65 + i];
        ssum[b * SEQ + s0 + n] = sum;
        __syncthreads();
    }
}

// ---------------------------------------------------------------------------
// y = s_sum[row] * x_inner * silu(res), rewritten in place as bf16 (GEMM4 A)
// ---------------------------------------------------------------------------
__global__ void ymul_kernel(__bf16* __restrict__ xinB,
                            const __bf16* __restrict__ gateB,
                            const float* __restrict__ ssum, size_t total)
{
    size_t i = (size_t)blockIdx.x * blockDim.x + threadIdx.x;
    if (i < total) {
        size_t row = i >> 11;   // / DINNER
        float y = ssum[row] * (float)xinB[i] * (float)gateB[i];
        xinB[i] = (__bf16)y;
    }
}

// ---------------------------------------------------------------------------
// Host orchestration
// ---------------------------------------------------------------------------
extern "C" void kernel_launch(void* const* d_in, const int* in_sizes, int n_in,
                              void* d_out, int out_size, void* d_ws, size_t ws_size,
                              hipStream_t stream) {
    const float* x     = (const float*)d_in[0];
    const float* W_in  = (const float*)d_in[1];
    const float* b_in  = (const float*)d_in[2];
    const float* W_xp  = (const float*)d_in[3];
    const float* b_xp  = (const float*)d_in[4];
    const float* W_dt  = (const float*)d_in[5];
    const float* b_dt  = (const float*)d_in[6];
    const float* W_out = (const float*)d_in[7];
    const float* b_out = (const float*)d_in[8];
    const float* A_log = (const float*)d_in[9];
    const float* Dvec  = (const float*)d_in[10];
    float* out = (float*)d_out;

    const size_t MB = 1048576;
    char* ws = (char*)d_ws;
    __bf16* xinB  = (__bf16*)(ws);                    // 32 MB
    __bf16* gateB = (__bf16*)(ws + 32 * MB);          // 32 MB
    __bf16* xB    = (__bf16*)(ws + 64 * MB);          // 16 MB  x as bf16
    __bf16* WtIn  = (__bf16*)(ws + 80 * MB);          //  8 MB  [4096,1024]
    __bf16* WtDt  = (__bf16*)(ws + 88 * MB);          //  4 MB  [1024,2048]
    __bf16* WtOut = (__bf16*)(ws + 92 * MB);          //  4 MB  [1024,2048]
    __bf16* WtXp  = (__bf16*)(ws + 96 * MB);          // 256 KB [64,2048]
    float*  bx    = (float*)(ws + 96 * MB + 262144);  //  2 MB
    float*  mdprt = (float*)(ws + 98 * MB + 262144);  // 256 KB
    float*  md    = (float*)(ws + 98 * MB + 524288);  // 32 KB
    float*  ssum  = (float*)(ws + 98 * MB + 557056);  // 32 KB

    // 0) one-time precision prep: x -> bf16, weights -> bf16 [N,K]
    {
        size_t n4 = (size_t)MROWS * DMODEL / 4;
        cvt_f32_bf16_kernel<<<(unsigned)((n4 + 255) / 256), 256, 0, stream>>>(x, xB, n4);
        dim3 gi(DMODEL / 32, (2 * DINNER) / 32);
        wtrans_kernel<<<gi, 256, 0, stream>>>(W_in, WtIn, DMODEL, 2 * DINNER);
        dim3 gd(DINNER / 32, DMODEL / 32);
        wtrans_kernel<<<gd, 256, 0, stream>>>(W_dt, WtDt, DINNER, DMODEL);
        wtrans_kernel<<<gd, 256, 0, stream>>>(W_out, WtOut, DINNER, DMODEL);
        dim3 gx(DINNER / 32, DSTATE / 32);
        wtrans_kernel<<<gx, 256, 0, stream>>>(W_xp, WtXp, DINNER, DSTATE);
    }
    // 1) in_proj + SiLU + split
    {
        dim3 g(MROWS / 128, (2 * DINNER) / 128);
        gemm_tile_kernel<0><<<g, 256, 0, stream>>>(
            xB, WtIn, b_in, DMODEL, 2 * DINNER,
            xinB, gateB, nullptr, nullptr, nullptr, nullptr);
    }
    // 2) delta GEMM + softplus + deterministic row partial sums
    {
        dim3 g(MROWS / 128, DMODEL / 128);
        gemm_tile_kernel<1><<<g, 256, 0, stream>>>(
            xinB, WtDt, b_dt, DINNER, DMODEL,
            nullptr, nullptr, mdprt, nullptr, nullptr, nullptr);
    }
    reduce_md_kernel<<<(MROWS + 255) / 256, 256, 0, stream>>>(mdprt, md);
    // 3) bx = xin @ W_xp + b_xp
    gemm_xp_kernel<<<MROWS / 128, 256, 0, stream>>>(xinB, WtXp, b_xp, bx);
    // 4) sequential selective scan (chunked, LDS-transposed reduction)
    scan_kernel<<<BATCH, 64, 0, stream>>>(bx, md, A_log, ssum);
    // 5) y = s_sum * x_inner * gate -> in-place bf16
    {
        size_t total = (size_t)MROWS * DINNER;
        ymul_kernel<<<(unsigned)((total + 255) / 256), 256, 0, stream>>>(
            xinB, gateB, ssum, total);
    }
    // 6) out = y @ W_out + b_out + x * D
    {
        dim3 g(MROWS / 128, DMODEL / 128);
        gemm_tile_kernel<2><<<g, 256, 0, stream>>>(
            xinB, WtOut, b_out, DINNER, DMODEL,
            nullptr, nullptr, nullptr, x, Dvec, out);
    }
}
